// GetPose_23484881175007
// MI455X (gfx1250) — compile-verified, matrix operand order checked by reading, not run on hardware
//
#include <hip/hip_runtime.h>
#include <hip/hip_bf16.h>

typedef __attribute__((ext_vector_type(16))) _Float16 v16h;
typedef __attribute__((ext_vector_type(8)))  _Float16 v8h;
typedef __attribute__((ext_vector_type(8)))  float    v8f;

#define NSAMP 92160   // 128*720 graphs = LSTM-input rows
#define TFRM  720
#define NB2   3840    // LSTM batch (sequences)
#define HDIM  128
#define G4    512     // 4*H gate width
#define MT    32      // batch rows per LSTM block
#define XPAD  64      // X row stride (K=36 zero-padded)

union AF { v16h v; _Float16 h[16]; v8h q[2]; };
union CF { v8f  v; float    f[8];  };
union HF { v8h  v; _Float16 h[8];  };

// A/B fragment k index for 16-bit 16x16x32 WMMA: per lane the 16 halfwords are
// two contiguous 8-half runs at k = 8*half and k = 16 + 8*half.
__device__ __forceinline__ int kidx(int j, int half) { return (j < 8 ? j : j + 8) + (half << 3); }
__device__ __forceinline__ float sigm(float x) { return 1.0f / (1.0f + __expf(-x)); }

// in-edge CSR of the 18-node skeleton graph (incl. self loops) + degrees
__constant__ int   c_off[19] = {0,3,7,10,12,13,16,18,19,21,23,24,26,28,29,31,33,34,35};
__constant__ int   c_src[35] = {0,15,14, 1,5,2,0, 2,3,8, 3,4, 4, 5,6,11, 6,7, 7,
                                8,9, 9,10, 10, 11,12, 12,13, 13, 14,16, 15,17, 16, 17};
__constant__ float c_deg[18] = {3,4,3,2,1,3,2,1,2,2,1,2,2,1,2,2,1,1};

// ---------------- Kernel 1: fused 2-layer GCN, one wave per graph sample ----
// Writes X in t-major row order r' = step*3840 + seq, stride 64, zero padded.
__global__ __launch_bounds__(256) void gcn_kernel(
    const float* __restrict__ pose, const float* __restrict__ W1,
    const float* __restrict__ b1,   const float* __restrict__ W2,
    const float* __restrict__ b2,   _Float16* __restrict__ X) {
  __shared__ float h1[8][18][24];
  const int wave = threadIdx.x >> 5, lane = threadIdx.x & 31;
  const int b = blockIdx.x * 8 + wave;          // sample index, grid exact
  const int n = b / TFRM, t = b % TFRM;
  const int seq = b / 24, step = b % 24;
  const size_t rowX = (size_t)(step * NB2 + seq) * XPAD;
  const float* px0 = pose + ((size_t)(n * 2 + 0) * TFRM + t) * 18;
  const float* px1 = px0 + (size_t)TFRM * 18;   // channel 1

  if (lane < 24) {                              // lane = hidden channel
    const float w0 = W1[lane], w1 = W1[24 + lane];
    float pre[18];
#pragma unroll
    for (int v = 0; v < 18; ++v) pre[v] = px0[v] * w0 + px1[v] * w1;
#pragma unroll
    for (int v = 0; v < 18; ++v) {
      float acc = b1[lane];
      const float dv = c_deg[v];
      for (int e = c_off[v]; e < c_off[v + 1]; ++e) {
        const int u = c_src[e];
        acc += __frsqrt_rn(c_deg[u] * dv) * pre[u];
      }
      h1[wave][v][lane] = fmaxf(acc, 0.0f);     // ReLU
    }
  }
  __syncthreads();
  if (lane < 18) {                              // lane = node, 2nd GCN layer
    const int v = lane;
    float o0 = b2[0], o1 = b2[1];
    const float dv = c_deg[v];
    for (int e = c_off[v]; e < c_off[v + 1]; ++e) {
      const int u = c_src[e];
      const float nrm = __frsqrt_rn(c_deg[u] * dv);
      float d0 = 0.0f, d1 = 0.0f;
#pragma unroll
      for (int c = 0; c < 24; ++c) {
        const float hv = h1[wave][u][c];
        d0 += hv * W2[c * 2 + 0];
        d1 += hv * W2[c * 2 + 1];
      }
      o0 += nrm * d0; o1 += nrm * d1;
    }
    X[rowX + v * 2 + 0] = (_Float16)o0;
    X[rowX + v * 2 + 1] = (_Float16)o1;
  } else {                                      // lanes 18..31 zero-pad k=36..63
    const int k = 36 + 2 * (lane - 18);
    X[rowX + k + 0] = (_Float16)0.0f;
    X[rowX + k + 1] = (_Float16)0.0f;
  }
}

// -------- Kernel 2: GX = X(92160x64) @ Wih^T + (bih+bhh), fragment-major ----
__global__ __launch_bounds__(256) void xgemm_kernel(
    const _Float16* __restrict__ X, const float* __restrict__ Wih,
    const float* __restrict__ bih,  const float* __restrict__ bhh,
    _Float16* __restrict__ GX) {
  extern __shared__ char smem[];
  _Float16* wihp = (_Float16*)smem;             // 2kt*32gt*32lane*16 = 64KB
  const int tid = threadIdx.x, lane = tid & 31;
  const int half = lane >> 4, ln = lane & 15;

  // Pack Wih^T (B[k][n] = Wih[n*36+k], K padded to 64) into B-frag layout
  for (int idx = tid; idx < 32768; idx += 256) {
    const int j = idx & 15, l = (idx >> 4) & 31, gt = (idx >> 9) & 31, kt = idx >> 14;
    const int k = kt * 32 + kidx(j, l >> 4);
    const int nn = gt * 16 + (l & 15);
    wihp[idx] = (k < 36) ? (_Float16)Wih[nn * 36 + k] : (_Float16)0.0f;
  }
  __syncthreads();

  const int wid = blockIdx.x * 8 + (tid >> 5);
  const int mt = wid >> 5, ntg = wid & 31;      // 5760 M-tiles x 32 N-tiles

  CF acc;
#pragma unroll
  for (int r = 0; r < 8; ++r) acc.f[r] = 0.0f;

  const _Float16* xrow = X + (size_t)(mt * 16 + ln) * XPAD;
#pragma unroll
  for (int kt = 0; kt < 2; ++kt) {
    AF a, bf;
    a.q[0] = *(const v8h*)(xrow + kt * 32 + 8 * half);        // global_load_b128
    a.q[1] = *(const v8h*)(xrow + kt * 32 + 16 + 8 * half);
    bf.v = *(const v16h*)(wihp + (((kt * 32 + ntg) * 32 + lane) << 4)); // 2x ds_load_b128
    acc.v = __builtin_amdgcn_wmma_f32_16x16x32_f16(false, a.v, false, bf.v,
                                                   (short)0, acc.v, false, false);
  }
  const float bias = bih[ntg * 16 + ln] + bhh[ntg * 16 + ln];
  HF o;
#pragma unroll
  for (int r = 0; r < 8; ++r) o.h[r] = (_Float16)(acc.f[r] + bias);
  // fragment-major: GX[((mt*32+gt)*32+lane)*8 + r]  -> one 16B store
  *(v8h*)(GX + (((size_t)(mt * 32 + ntg) * 32 + lane) << 3)) = o.v;
}

// -------- Kernel 3: persistent LSTM, Whh^T pre-packed in LDS B-frag layout --
__global__ __launch_bounds__(256) void lstm_kernel(
    const _Float16* __restrict__ GX, const float* __restrict__ Whh,
    float* __restrict__ out) {
  extern __shared__ char smem[];
  _Float16* whhp = (_Float16*)smem;                       // 4kt*32gt*32lane*16 = 128KB
  _Float16* hbuf = (_Float16*)(smem + 65536 * 2);         // 2 x MT x 128 f16 = 16KB
  const int tid = threadIdx.x, wave = tid >> 5, lane = tid & 31;
  const int half = lane >> 4, ln = lane & 15;
  const int msub = wave & 1;                              // 2 m-subtiles of 16 rows
  const int ngrp = wave >> 1;                             // 4 groups x 2 h n-tiles
  const int s0 = blockIdx.x * MT;
  const int mtb = s0 >> 4;                                // base m-tile (in 16-row units)

  // Pack Whh^T (B[k][n] = Whh[n*128+k]) into per-lane-contiguous fragments
  for (int idx = tid; idx < 65536; idx += 256) {
    const int j = idx & 15, l = (idx >> 4) & 31, gt = (idx >> 9) & 31, kt = idx >> 14;
    const int k = kt * 32 + kidx(j, l >> 4);
    const int nn = gt * 16 + (l & 15);
    whhp[idx] = (_Float16)Whh[nn * HDIM + k];
  }
  for (int idx = tid; idx < MT * HDIM; idx += 256) hbuf[idx] = (_Float16)0.0f; // h0 = 0
  __syncthreads();

  float creg[2][8];
#pragma unroll
  for (int i = 0; i < 2; ++i)
#pragma unroll
    for (int r = 0; r < 8; ++r) creg[i][r] = 0.0f;

  for (int t = 0; t < 24; ++t) {
    __syncthreads();                                      // prev-step h visible
    const _Float16* hr = hbuf + (t & 1) * MT * HDIM;
    _Float16*       hw = hbuf + ((t + 1) & 1) * MT * HDIM;

    if (t < 23)                                           // prefetch next step's 32KB GX region
      __builtin_prefetch(GX + ((size_t)((t + 1) * 240 + mtb) << 13) + (size_t)tid * 64, 0, 1);

    AF a[4];                                              // h A-frags, reused for all gates/ntiles
    const _Float16* hrow = hr + (msub * 16 + ln) * HDIM;
#pragma unroll
    for (int kt = 0; kt < 4; ++kt) {
      a[kt].q[0] = *(const v8h*)(hrow + kt * 32 + 8 * half);       // ds_load_b128
      a[kt].q[1] = *(const v8h*)(hrow + kt * 32 + 16 + 8 * half);
    }

    const int mtile = t * 240 + mtb + msub;               // this wave's GX m-tile
#pragma unroll
    for (int ntl = 0; ntl < 2; ++ntl) {
      const int nt = ngrp * 2 + ntl;                      // h column tile 0..7
      CF acc[4];
#pragma unroll
      for (int g = 0; g < 4; ++g) {                       // init accum = GX (x-part + bias)
        HF gx;
        gx.v = *(const v8h*)(GX + (((size_t)(mtile * 32 + g * 8 + nt) * 32 + lane) << 3));
#pragma unroll
        for (int r = 0; r < 8; ++r) acc[g].f[r] = (float)gx.h[r];
      }
#pragma unroll
      for (int kt = 0; kt < 4; ++kt)
#pragma unroll
        for (int g = 0; g < 4; ++g) {
          const int gt = g * 8 + nt;
          AF bf;
          bf.v = *(const v16h*)(whhp + (((kt * 32 + gt) * 32 + lane) << 4)); // 2x ds_load_b128
          acc[g].v = __builtin_amdgcn_wmma_f32_16x16x32_f16(
              false, a[kt].v, false, bf.v, (short)0, acc[g].v, false, false);
        }
#pragma unroll
      for (int r = 0; r < 8; ++r) {                       // gates: i,f,g,o
        const float cc = sigm(acc[1].f[r]) * creg[ntl][r] +
                         sigm(acc[0].f[r]) * tanhf(acc[2].f[r]);
        creg[ntl][r] = cc;
        const float hh = sigm(acc[3].f[r]) * tanhf(cc);
        const int m = msub * 16 + r + 8 * half;
        hw[m * HDIM + nt * 16 + ln] = (_Float16)hh;
        out[(size_t)(s0 + m) * 24 * HDIM + t * HDIM + nt * 16 + ln] = hh;
      }
    }
  }
}

extern "C" void kernel_launch(void* const* d_in, const int* in_sizes, int n_in,
                              void* d_out, int out_size, void* d_ws, size_t ws_size,
                              hipStream_t stream) {
  const float* pose = (const float*)d_in[0];
  const float* W1   = (const float*)d_in[1];
  const float* b1   = (const float*)d_in[2];
  const float* W2   = (const float*)d_in[3];
  const float* b2   = (const float*)d_in[4];
  const float* Wih  = (const float*)d_in[5];
  const float* Whh  = (const float*)d_in[6];
  const float* bih  = (const float*)d_in[7];
  const float* bhh  = (const float*)d_in[8];
  float* out = (float*)d_out;

  _Float16* X  = (_Float16*)d_ws;                                   // 92160*64 f16
  _Float16* GX = (_Float16*)((char*)d_ws + (size_t)NSAMP * XPAD * 2); // 92160*512 f16, frag-major

  const int lds_lstm  = 65536 * 2 + 2 * MT * HDIM * 2;              // 147456
  const int lds_xgemm = 32768 * 2;                                  // 65536
  (void)hipFuncSetAttribute((const void*)lstm_kernel,
                            hipFuncAttributeMaxDynamicSharedMemorySize, lds_lstm);
  (void)hipFuncSetAttribute((const void*)xgemm_kernel,
                            hipFuncAttributeMaxDynamicSharedMemorySize, lds_xgemm);

  gcn_kernel<<<NSAMP / 8, 256, 0, stream>>>(pose, W1, b1, W2, b2, X);
  xgemm_kernel<<<(NSAMP / 16) * 32 / 8, 256, lds_xgemm, stream>>>(X, Wih, bih, bhh, GX);
  lstm_kernel<<<NB2 / MT, 256, lds_lstm, stream>>>(GX, Whh, out);
}
